// AutoregressiveEncoder_25065429139511
// MI455X (gfx1250) — matrix operand order repaired and checked by reading, not compiled, not run on hardware
//
#include <hip/hip_runtime.h>
#include <hip/hip_bf16.h>

#define B_        32
#define T_        512
#define IN_       512
#define OUT_      512
#define LH_       512
#define LSTM_IN_  1024
#define KCAT      1536   // LSTM_IN + LH (fused ih|hh)
#define GATES     2048   // 4*LH
#define NWG       64
#define TPB       256    // 8 wave32s per WG -> 512 waves total

typedef __attribute__((ext_vector_type(16))) __bf16   v16bf;
typedef __attribute__((ext_vector_type(8)))  float    v8f;
typedef __attribute__((ext_vector_type(4)))  unsigned u32x4;

union ABFrag { v16bf v; u32x4 q[2]; };

__device__ __forceinline__ unsigned short f2bf(float f) {
  unsigned u = __builtin_bit_cast(unsigned, f);
  u += 0x7FFFu + ((u >> 16) & 1u);          // round-to-nearest-even
  return (unsigned short)(u >> 16);
}
__device__ __forceinline__ float sigmf(float v) { return 1.f / (1.f + __expf(-v)); }

// ---------------- device-wide barrier (generation counter) ----------------
__device__ __forceinline__ void gbar(unsigned* bar, unsigned* gen) {
  __syncthreads();
  if (threadIdx.x == 0) {
    unsigned s = *gen + 1u;
    *gen = s;
    unsigned prev = __hip_atomic_fetch_add(&bar[0], 1u, __ATOMIC_ACQ_REL,
                                           __HIP_MEMORY_SCOPE_AGENT);
    if (prev == NWG - 1u) {
      __hip_atomic_store(&bar[0], 0u, __ATOMIC_RELAXED, __HIP_MEMORY_SCOPE_AGENT);
      __hip_atomic_store(&bar[1], s,  __ATOMIC_RELEASE, __HIP_MEMORY_SCOPE_AGENT);
    } else {
      while (__hip_atomic_load(&bar[1], __ATOMIC_ACQUIRE,
                               __HIP_MEMORY_SCOPE_AGENT) < s)
        __builtin_amdgcn_s_sleep(1);
    }
  }
  __syncthreads();
}

// ---------------- gate GEMM: gates[cell] = zb[cell] @ Wt[cell] ----------------
// 512 waves, one 16x16 tile each: cell(2) x mtile(2) x ntile(128), K = 1536.
__device__ __forceinline__ void gemm_gates(const unsigned short* __restrict__ zb,
                                           const unsigned short* __restrict__ Wt,
                                           float* __restrict__ gates,
                                           int wave, int lane, int cellBase) {
  const int cell  = cellBase + (wave >> 8);
  const int rem   = wave & 255;
  const int mtile = rem >> 7;
  const int ntile = rem & 127;
  const int hi    = (lane >> 4) * 8;                 // K-phase / M-row offset
  const unsigned short* Abase = zb + ((size_t)cell * B_ + mtile * 16 + (lane & 15)) * KCAT;
  const unsigned short* Bbase = Wt + (size_t)cell * KCAT * GATES + ntile * 16 + (lane & 15);
  v8f acc = {0.f,0.f,0.f,0.f,0.f,0.f,0.f,0.f};
  for (int k0 = 0; k0 < KCAT; k0 += 32) {
    ABFrag a, b;
    a.q[0] = *(const u32x4*)(Abase + k0 + hi);
    a.q[1] = *(const u32x4*)(Abase + k0 + hi + 16);
    const unsigned short* bp = Bbase + (size_t)k0 * GATES;
    #pragma unroll
    for (int s = 0; s < 8; ++s) {
      b.v[s]     = __builtin_bit_cast(__bf16, bp[(size_t)(hi + s) * GATES]);
      b.v[s + 8] = __builtin_bit_cast(__bf16, bp[(size_t)(16 + hi + s) * GATES]);
    }
    acc = __builtin_amdgcn_wmma_f32_16x16x32_bf16(false, a.v, false, b.v,
                                                  (short)0, acc, false, false);
  }
  const int n = ntile * 16 + (lane & 15);
  float* g = gates + (size_t)cell * B_ * GATES + n;
  const int mb = mtile * 16 + hi;                    // lanes>=16 hold rows r+8
  #pragma unroll
  for (int r = 0; r < 8; ++r) g[(size_t)(mb + r) * GATES] = acc[r];
}

// ---------------- elementwise LSTM cell update ----------------
__device__ __forceinline__ void cell_update(const float* __restrict__ gates,
                                            const float* __restrict__ bsum,
                                            float* __restrict__ hs,
                                            float* __restrict__ cs,
                                            unsigned short* __restrict__ zb,
                                            unsigned short* __restrict__ fcin,
                                            int tid, int layer) {
  const int c0 = layer * 2;
  for (int i = tid; i < 2 * B_ * LH_; i += NWG * TPB) {
    const int cell = c0 + (i / (B_ * LH_));
    const int rem  = i % (B_ * LH_);
    const int m = rem / LH_;
    const int n = rem % LH_;
    const float* g  = gates + ((size_t)cell * B_ + m) * GATES;
    const float* bs = bsum + (size_t)cell * GATES;
    const float gi = g[n]           + bs[n];
    const float gf = g[LH_ + n]     + bs[LH_ + n];
    const float gc = g[2 * LH_ + n] + bs[2 * LH_ + n];
    const float go = g[3 * LH_ + n] + bs[3 * LH_ + n];
    const size_t si = ((size_t)cell * B_ + m) * LH_ + n;
    const float cn = sigmf(gf) * cs[si] + sigmf(gi) * tanhf(gc);
    const float hn = sigmf(go) * tanhf(cn);
    cs[si] = cn;
    hs[si] = hn;
    const unsigned short hb = f2bf(hn);
    zb[((size_t)cell * B_ + m) * KCAT + LSTM_IN_ + n] = hb;  // own recurrent h
    const int d = cell & 1;
    if (layer == 0) {                                        // next layer input
      zb[((size_t)2 * B_ + m) * KCAT + d * LH_ + n] = hb;
      zb[((size_t)3 * B_ + m) * KCAT + d * LH_ + n] = hb;
    } else {
      fcin[(size_t)m * LSTM_IN_ + d * LH_ + n] = hb;         // FC input
    }
  }
}

// ---------------- FC head + prev feedback + x(t+1) prefetch ----------------
__device__ __forceinline__ void fc_phase(const unsigned short* __restrict__ fcin,
                                         const unsigned short* __restrict__ WfcT,
                                         const float* __restrict__ bfc,
                                         const float* __restrict__ x,
                                         unsigned short* __restrict__ zb,
                                         float* __restrict__ out,
                                         int wave, int lane, int tid, int t) {
  if (wave < 64) {                    // 2 mtiles x 32 ntiles, K = 1024
    const int mtile = wave >> 5;
    const int ntile = wave & 31;
    const int hi = (lane >> 4) * 8;
    const unsigned short* Abase = fcin + (size_t)(mtile * 16 + (lane & 15)) * LSTM_IN_;
    const unsigned short* Bbase = WfcT + ntile * 16 + (lane & 15);
    v8f acc = {0.f,0.f,0.f,0.f,0.f,0.f,0.f,0.f};
    for (int k0 = 0; k0 < LSTM_IN_; k0 += 32) {
      ABFrag a, b;
      a.q[0] = *(const u32x4*)(Abase + k0 + hi);
      a.q[1] = *(const u32x4*)(Abase + k0 + hi + 16);
      const unsigned short* bp = Bbase + (size_t)k0 * OUT_;
      #pragma unroll
      for (int s = 0; s < 8; ++s) {
        b.v[s]     = __builtin_bit_cast(__bf16, bp[(size_t)(hi + s) * OUT_]);
        b.v[s + 8] = __builtin_bit_cast(__bf16, bp[(size_t)(16 + hi + s) * OUT_]);
      }
      acc = __builtin_amdgcn_wmma_f32_16x16x32_bf16(false, a.v, false, b.v,
                                                    (short)0, acc, false, false);
    }
    const int n = ntile * 16 + (lane & 15);
    const float bv = bfc[n];
    #pragma unroll
    for (int r = 0; r < 8; ++r) {
      const int m = mtile * 16 + hi + r;
      const float val = acc[r] + bv;
      out[((size_t)m * T_ + t) * OUT_ + n] = val;            // outputs[B][T][OUT]
      const unsigned short pb = f2bf(val);                   // prev = detach(pred)
      zb[((size_t)0 * B_ + m) * KCAT + IN_ + n] = pb;
      zb[((size_t)1 * B_ + m) * KCAT + IN_ + n] = pb;
    }
  } else if (t + 1 < T_) {            // remaining waves stage x(t+1)
    const int idx = tid - 64 * 32;
    for (int i = idx; i < B_ * IN_; i += NWG * TPB - 64 * 32) {
      const int m = i >> 9;
      const int k = i & (IN_ - 1);
      const unsigned short v = f2bf(x[((size_t)m * T_ + (t + 1)) * IN_ + k]);
      zb[((size_t)0 * B_ + m) * KCAT + k] = v;
      zb[((size_t)1 * B_ + m) * KCAT + k] = v;
    }
  }
}

// ---------------- persistent kernel: all 512 timesteps ----------------
__global__ __launch_bounds__(TPB, 1) void lstm_persistent(
    const float* __restrict__ x,
    const unsigned short* __restrict__ Wt,
    const unsigned short* __restrict__ WfcT,
    const float* __restrict__ bsum,
    const float* __restrict__ bfc,
    unsigned short* __restrict__ zb,
    unsigned short* __restrict__ fcin,
    float* __restrict__ gates,
    float* __restrict__ hs,
    float* __restrict__ cs,
    unsigned* __restrict__ bar,
    float* __restrict__ out) {
  __shared__ unsigned gen;
  if (threadIdx.x == 0) gen = 0;
  __syncthreads();
  const int tid  = blockIdx.x * TPB + threadIdx.x;
  const int lane = threadIdx.x & 31;
  const int wave = tid >> 5;

  // stage x_0 (prev and h parts of zb were zeroed by the init kernel)
  for (int i = tid; i < B_ * IN_; i += NWG * TPB) {
    const int m = i >> 9, k = i & (IN_ - 1);
    const unsigned short v = f2bf(x[(size_t)m * T_ * IN_ + k]);
    zb[((size_t)0 * B_ + m) * KCAT + k] = v;
    zb[((size_t)1 * B_ + m) * KCAT + k] = v;
  }
  gbar(bar, &gen);

  for (int t = 0; t < T_; ++t) {
    gemm_gates(zb, Wt, gates, wave, lane, 0);                 // layer 0, both dirs
    gbar(bar, &gen);
    cell_update(gates, bsum, hs, cs, zb, fcin, tid, 0);
    gbar(bar, &gen);
    gemm_gates(zb, Wt, gates, wave, lane, 2);                 // layer 1, both dirs
    gbar(bar, &gen);
    cell_update(gates, bsum, hs, cs, zb, fcin, tid, 1);
    gbar(bar, &gen);
    fc_phase(fcin, WfcT, bfc, x, zb, out, wave, lane, tid, t);
    gbar(bar, &gen);
  }

  const size_t OUTS = (size_t)B_ * T_ * OUT_;                 // then h_n, c_n
  for (int i = tid; i < 4 * B_ * LH_; i += NWG * TPB) {
    out[OUTS + i] = hs[i];
    out[OUTS + 4 * B_ * LH_ + i] = cs[i];
  }
}

// ---------------- init kernels ----------------
__global__ void prep_weights(const float* __restrict__ Wih, const float* __restrict__ Whh,
                             const float* __restrict__ bih, const float* __restrict__ bhh,
                             const float* __restrict__ Wfc,
                             unsigned short* __restrict__ Wt,
                             unsigned short* __restrict__ WfcT,
                             float* __restrict__ bsum) {
  const size_t WT_E  = 4UL * KCAT * GATES;
  const size_t WFC_E = (size_t)LSTM_IN_ * OUT_;
  const size_t BS_E  = 4UL * GATES;
  for (size_t i = (size_t)blockIdx.x * blockDim.x + threadIdx.x;
       i < WT_E + WFC_E + BS_E; i += (size_t)gridDim.x * blockDim.x) {
    if (i < WT_E) {
      const size_t cell = i / ((size_t)KCAT * GATES);
      const size_t r    = i % ((size_t)KCAT * GATES);
      const size_t k = r / GATES, n = r % GATES;
      const float v = (k < LSTM_IN_)
          ? Wih[(cell * GATES + n) * LSTM_IN_ + k]
          : Whh[(cell * GATES + n) * LH_ + (k - LSTM_IN_)];
      Wt[i] = f2bf(v);                      // Wt[cell][k][n] (pre-transposed)
    } else if (i < WT_E + WFC_E) {
      const size_t j = i - WT_E;
      const size_t k = j / OUT_, n = j % OUT_;
      WfcT[j] = f2bf(Wfc[n * (size_t)LSTM_IN_ + k]);   // W_fc is [OUT][HID]
    } else {
      const size_t j = i - WT_E - WFC_E;
      bsum[j] = bih[j] + bhh[j];
    }
  }
}

__global__ void zero_ws(u32x4* __restrict__ p, size_t n16) {
  for (size_t i = (size_t)blockIdx.x * blockDim.x + threadIdx.x; i < n16;
       i += (size_t)gridDim.x * blockDim.x) {
    u32x4 z = {0u, 0u, 0u, 0u};
    p[i] = z;
  }
}

// ---------------- launch ----------------
extern "C" void kernel_launch(void* const* d_in, const int* in_sizes, int n_in,
                              void* d_out, int out_size, void* d_ws, size_t ws_size,
                              hipStream_t stream) {
  (void)in_sizes; (void)n_in; (void)out_size; (void)ws_size;
  const float* x   = (const float*)d_in[0];
  // d_in[1] = input_lengths (all == T, unused)
  const float* Wih = (const float*)d_in[2];
  const float* Whh = (const float*)d_in[3];
  const float* bih = (const float*)d_in[4];
  const float* bhh = (const float*)d_in[5];
  const float* Wfc = (const float*)d_in[6];
  const float* bfc = (const float*)d_in[7];
  float* out = (float*)d_out;
  char*  ws  = (char*)d_ws;

  constexpr size_t OFF_WT    = 0;
  constexpr size_t OFF_WFC   = OFF_WT   + 4UL * KCAT * GATES * 2;     // 25.2 MB bf16
  constexpr size_t OFF_BSUM  = OFF_WFC  + (size_t)LSTM_IN_ * OUT_ * 2;
  constexpr size_t OFF_ZB    = OFF_BSUM + 4UL * GATES * 4;
  constexpr size_t OFF_FCIN  = OFF_ZB   + 4UL * B_ * KCAT * 2;
  constexpr size_t OFF_HS    = OFF_FCIN + (size_t)B_ * LSTM_IN_ * 2;
  constexpr size_t OFF_CS    = OFF_HS   + 4UL * B_ * LH_ * 4;
  constexpr size_t OFF_BAR   = OFF_CS   + 4UL * B_ * LH_ * 4;
  constexpr size_t OFF_GATES = OFF_BAR  + 256;

  unsigned short* Wt    = (unsigned short*)(ws + OFF_WT);
  unsigned short* WfcT  = (unsigned short*)(ws + OFF_WFC);
  float*          bsum  = (float*)(ws + OFF_BSUM);
  unsigned short* zb    = (unsigned short*)(ws + OFF_ZB);
  unsigned short* fcin  = (unsigned short*)(ws + OFF_FCIN);
  float*          hs    = (float*)(ws + OFF_HS);
  float*          cs    = (float*)(ws + OFF_CS);
  unsigned*       bar   = (unsigned*)(ws + OFF_BAR);
  float*          gates = (float*)(ws + OFF_GATES);

  zero_ws<<<256, 256, 0, stream>>>((u32x4*)(ws + OFF_ZB), (OFF_GATES - OFF_ZB) / 16);
  prep_weights<<<8192, 256, 0, stream>>>(Wih, Whh, bih, bhh, Wfc, Wt, WfcT, bsum);
  lstm_persistent<<<NWG, TPB, 0, stream>>>(x, Wt, WfcT, bsum, bfc, zb, fcin,
                                           gates, hs, cs, bar, out);
}
